// CNNTransBlock_27479200759864
// MI455X (gfx1250) — matrix operand order repaired and checked by reading, not compiled, not run on hardware
//
#include <hip/hip_runtime.h>
#include <math.h>

typedef _Float16 half_t;
typedef __attribute__((ext_vector_type(16))) _Float16 v16h;
typedef __attribute__((ext_vector_type(8)))  _Float16 h8;
typedef __attribute__((ext_vector_type(8)))  float    v8f;
typedef __attribute__((ext_vector_type(4))) unsigned int uint4v;
typedef __attribute__((ext_vector_type(8))) int int8v;
typedef __attribute__((ext_vector_type(4))) int int4v;

#define BB   16
#define SS   1024
#define DD   128
#define HH   4
#define DHH  32
#define KT   31
#define PADK 15
#define EPSF 1e-5f

#if defined(__has_builtin)
#if __has_builtin(__builtin_amdgcn_tensor_load_to_lds) && __has_builtin(__builtin_amdgcn_s_wait_tensorcnt)
#define HAS_TDM 1
#endif
#endif

// ---------------------------------------------------------------- WMMA helpers
static __device__ __forceinline__ v8f wmma_f16(v16h a, v16h b, v8f c) {
  // D = A(16x32 f16) * B(32x16 f16) + C(16x16 f32)
  return __builtin_amdgcn_wmma_f32_16x16x32_f16(false, a, false, b, (short)0, c,
                                                false, false);
}

static __device__ __forceinline__ v16h frag_zero() {
  v16h f;
#pragma unroll
  for (int i = 0; i < 16; ++i) f[i] = (half_t)0.f;
  return f;
}

// A fragment (16x32, row-major source, row stride ld halves).
// ISA layout: lanes 0-15 = rows 0-15 with K {0..7,16..23}; lanes 16-31 same rows
// with K {8..15,24..31}.
static __device__ __forceinline__ v16h frag_load_a(const half_t* p, int ld) {
  int lane = threadIdx.x & 31;
  int r    = lane & 15;
  int k0   = (lane >> 4) << 3;  // 0 or 8
  const half_t* src = p + (size_t)r * ld + k0;
  h8 lo8 = *(const h8*)(src);
  h8 hi8 = *(const h8*)(src + 16);
  v16h f;
#pragma unroll
  for (int i = 0; i < 8; ++i) { f[i] = lo8[i]; f[8 + i] = hi8[i]; }
  return f;
}

// Guarded A fragment: row r valid iff 0 <= s0+r < slen (for conv taps / +1 row)
static __device__ __forceinline__ v16h frag_load_a_guard(const half_t* p, int ld,
                                                         int s0, int slen) {
  int lane = threadIdx.x & 31;
  int r    = lane & 15;
  int k0   = (lane >> 4) << 3;
  int sr   = s0 + r;
  if ((unsigned)sr < (unsigned)slen) {
    const half_t* src = p + (size_t)r * ld + k0;
    h8 lo8 = *(const h8*)(src);
    h8 hi8 = *(const h8*)(src + 16);
    v16h f;
#pragma unroll
    for (int i = 0; i < 8; ++i) { f[i] = lo8[i]; f[8 + i] = hi8[i]; }
    return f;
  }
  return frag_zero();
}

// B fragment (32x16). Source is B^T row-major [N][K] with row stride ld.
// ISA layout: lane holds column N=lane&15; lanes 0-15 K {0..7,16..23}; lanes
// 16-31 K {8..15,24..31}.
static __device__ __forceinline__ v16h frag_load_b(const half_t* p, int ld) {
  int lane = threadIdx.x & 31;
  int n    = lane & 15;
  int k0   = (lane >> 4) << 3;  // 0 or 8
  const half_t* src = p + (size_t)n * ld + k0;
  h8 lo8 = *(const h8*)(src);
  h8 hi8 = *(const h8*)(src + 16);
  v16h f;
#pragma unroll
  for (int i = 0; i < 8; ++i) { f[i] = lo8[i]; f[8 + i] = hi8[i]; }
  return f;
}

#if defined(HAS_TDM)
// Issue a TDM 2D tile load: rows x rowlen f16 elements, row-major, contiguous
// (tensor_dim0_stride == rowlen), global -> LDS. Descriptor packing per
// CDNA5 ISA (D# groups 0/1; remaining groups zero for 2D
// tiles). This toolchain's builtin takes 6 args (extra int32x8 group + cpol).
static __device__ __forceinline__ void tdm_load_2d(const half_t* gsrc, half_t* lds,
                                                   int rows, int rowlen) {
  unsigned long long ga = (unsigned long long)(uintptr_t)gsrc;
  unsigned int la = (unsigned int)(uintptr_t)lds;  // low 32 bits = LDS byte offset
  uint4v g0;
  g0[0] = 1u;                                        // count=1 (valid user D#)
  g0[1] = la;                                        // lds_addr
  g0[2] = (unsigned int)(ga & 0xffffffffu);          // global_addr[31:0]
  g0[3] = (unsigned int)((ga >> 32) & 0x01ffffffu)   // global_addr[56:32]
          | (2u << 30);                              // type=2 ("image")
  unsigned int dim0 = (unsigned int)rowlen;          // tensor_dim0 (elems)
  unsigned int dim1 = (unsigned int)rows;            // tensor_dim1
  unsigned int st0  = (unsigned int)rowlen;          // tensor_dim0_stride (elems)
  int8v g1;
  g1[0] = (int)(1u << 16);                           // wg_mask=0, data_size=1 (2B)
  g1[1] = (int)((dim0 & 0xffffu) << 16);             // tensor_dim0[15:0] @bit48
  g1[2] = (int)((dim0 >> 16) | ((dim1 & 0xffffu) << 16));
  g1[3] = (int)((dim1 >> 16) | ((dim0 & 0xffffu) << 16)); // tile_dim0 @bit112
  g1[4] = (int)(dim1 & 0xffffu);                     // tile_dim1, tile_dim2=0
  g1[5] = (int)st0;                                  // tensor_dim0_stride[31:0]
  g1[6] = 0;                                         // stride[47:32]=0, d1stride=0
  g1[7] = 0;
  int4v g2; g2[0] = 0; g2[1] = 0; g2[2] = 0; g2[3] = 0;
  int4v g3; g3[0] = 0; g3[1] = 0; g3[2] = 0; g3[3] = 0;
  int8v g4;
#pragma unroll
  for (int i = 0; i < 8; ++i) g4[i] = 0;
  __builtin_amdgcn_tensor_load_to_lds(g0, g1, g2, g3, g4, 0);
}
#endif

// ---------------------------------------------------------------- elementwise
__global__ void f32_to_f16_kernel(const float* __restrict__ in,
                                  half_t* __restrict__ out, int n) {
  int i = blockIdx.x * 256 + threadIdx.x;
  if (i < n) out[i] = (half_t)in[i];
}

// [O][I][Kt] f32 -> [Kt][O][I] f16
__global__ void pack_conv_kernel(const float* __restrict__ in,
                                 half_t* __restrict__ out, int O, int I, int Kt) {
  int idx = blockIdx.x * 256 + threadIdx.x;
  int n = O * I * Kt;
  if (idx >= n) return;
  int o = idx / (I * Kt);
  int r = idx - o * I * Kt;
  int i = r / Kt;
  int k = r - i * Kt;
  out[((size_t)k * O + o) * I + i] = (half_t)in[idx];
}

__global__ void pe_kernel(half_t* __restrict__ peh) {
  int idx = blockIdx.x * 256 + threadIdx.x;
  if (idx >= SS * DD) return;
  int s  = idx >> 7;
  int c  = idx & 127;
  int i2 = c >> 1;
  float dv = __expf((float)(2 * i2) * (-9.210340371976184f / 128.f));
  float a  = (float)s * dv;
  peh[idx] = (half_t)((c & 1) ? cosf(a) : sinf(a));
}

// per-row LayerNorm over D=128 (one wave per row); optional per-batch prescale
__global__ void __launch_bounds__(256)
rowln_kernel(const float* __restrict__ in, const float* __restrict__ gw,
             const float* __restrict__ gb, const float* __restrict__ prestats,
             half_t* __restrict__ outh) {
  int wave = threadIdx.x >> 5, lane = threadIdx.x & 31;
  int row = blockIdx.x * 8 + wave;
  const float* src = in + (size_t)row * DD + lane * 4;
  float4 xv = *(const float4*)src;
  float x[4] = {xv.x, xv.y, xv.z, xv.w};
  if (prestats) {
    int b = row / SS;
    float m = prestats[2 * b], rs = prestats[2 * b + 1];
#pragma unroll
    for (int t = 0; t < 4; ++t) x[t] = (x[t] - m) * rs;
  }
  float s = x[0] + x[1] + x[2] + x[3];
  float s2 = x[0]*x[0] + x[1]*x[1] + x[2]*x[2] + x[3]*x[3];
#pragma unroll
  for (int off = 1; off < 32; off <<= 1) {
    s  += __shfl_xor(s, off);
    s2 += __shfl_xor(s2, off);
  }
  float mean = s * (1.f / DD);
  float var  = s2 * (1.f / DD) - mean * mean;
  float rstd = rsqrtf(var + EPSF);
#pragma unroll
  for (int t = 0; t < 4; ++t) {
    int c = lane * 4 + t;
    outh[(size_t)row * DD + c] = (half_t)((x[t] - mean) * rstd * gw[c] + gb[c]);
  }
}

// ---------------------------------------------------------------- GEMM / conv
// out[m, n] = sum_t sum_k A[m + (t-pad) (in s), k] * W[t][n][k]  (+bias)(+act)(+resid)
// A: [M, cin] f16 (rows grouped per batch of slen), W: [taps][cout][cin] f16.
// Per tap, wave 0 TDM-loads the block's 64xCin weight tile to LDS; all 8 waves
// read B fragments from LDS.
__global__ void __launch_bounds__(256)
gemm_conv_kernel(const half_t* __restrict__ A, const half_t* __restrict__ W,
                 const float* __restrict__ bias, const float* __restrict__ resid,
                 float* __restrict__ outF, half_t* __restrict__ outH,
                 int slen, int cin, int cout, int taps, int pad, int act) {
  extern __shared__ char smem_raw[];
  half_t* wlds = (half_t*)smem_raw;  // [64][cin] f16 weight tile
  int wave = threadIdx.x >> 5, lane = threadIdx.x & 31;
  int row0 = blockIdx.x * 128 + (wave >> 1) * 32;
  int colbase = blockIdx.y * 64;
  int wcol = (wave & 1) * 32;          // column offset within LDS tile
  int col0 = colbase + wcol;
  v8f acc00, acc01, acc10, acc11;
#pragma unroll
  for (int j = 0; j < 8; ++j) { acc00[j] = 0.f; acc01[j] = 0.f; acc10[j] = 0.f; acc11[j] = 0.f; }
  int s0 = row0 % slen;
  for (int t = 0; t < taps; ++t) {
    const half_t* Wtap = W + ((size_t)t * cout + colbase) * cin;
#if defined(HAS_TDM)
    if (wave == 0) {
      tdm_load_2d(Wtap, wlds, 64, cin);
      __builtin_amdgcn_s_wait_tensorcnt(0);
    }
#else
    {
      const h8* src = (const h8*)Wtap;
      h8* dst = (h8*)wlds;
      int total = (64 * cin) >> 3;
      for (int i = threadIdx.x; i < total; i += 256) dst[i] = src[i];
    }
#endif
    __syncthreads();

    int shift = t - pad;
    int sA = s0 + shift;
    const half_t* Abase = A + (size_t)(row0 + shift) * cin;
    if (t + 1 < taps) {  // prefetch next tap's activation rows
      __builtin_prefetch(A + (size_t)(row0 + shift + 1) * cin, 0, 1);
    }
    bool full = (sA >= 0) && (sA + 31 < slen);   // wave-uniform fast path
    if (full) {
      for (int kk = 0; kk < cin; kk += 32) {
        v16h a0 = frag_load_a(Abase + kk, cin);
        v16h a1 = frag_load_a(Abase + (size_t)16 * cin + kk, cin);
        v16h b0 = frag_load_b(wlds + (size_t)wcol * cin + kk, cin);
        v16h b1 = frag_load_b(wlds + (size_t)(wcol + 16) * cin + kk, cin);
        acc00 = wmma_f16(a0, b0, acc00);
        acc01 = wmma_f16(a0, b1, acc01);
        acc10 = wmma_f16(a1, b0, acc10);
        acc11 = wmma_f16(a1, b1, acc11);
      }
    } else {
      for (int kk = 0; kk < cin; kk += 32) {
        v16h a0 = frag_load_a_guard(Abase + kk, cin, sA, slen);
        v16h a1 = frag_load_a_guard(Abase + (size_t)16 * cin + kk, cin, sA + 16, slen);
        v16h b0 = frag_load_b(wlds + (size_t)wcol * cin + kk, cin);
        v16h b1 = frag_load_b(wlds + (size_t)(wcol + 16) * cin + kk, cin);
        acc00 = wmma_f16(a0, b0, acc00);
        acc01 = wmma_f16(a0, b1, acc01);
        acc10 = wmma_f16(a1, b0, acc10);
        acc11 = wmma_f16(a1, b1, acc11);
      }
    }
    __syncthreads();  // protect LDS tile before next tap overwrites it
  }
  int hi = lane >> 4, cl = lane & 15;
#pragma unroll
  for (int i = 0; i < 2; ++i) {
#pragma unroll
    for (int jj = 0; jj < 2; ++jj) {
      v8f acc = (i == 0) ? (jj == 0 ? acc00 : acc01) : (jj == 0 ? acc10 : acc11);
      int col = col0 + jj * 16 + cl;
      float bvv = bias ? bias[col] : 0.f;
#pragma unroll
      for (int j = 0; j < 8; ++j) {
        int row = row0 + i * 16 + j + 8 * hi;
        float v = acc[j] + bvv;
        if (act == 1) v = v / (1.f + __expf(-v));  // swish
        if (resid) v += resid[(size_t)row * cout + col];
        size_t o = (size_t)row * cout + col;
        if (outF) outF[o] = v;
        if (outH) outH[o] = (half_t)v;
      }
    }
  }
}

// ---------------------------------------------------------------- attention
__global__ void qkv_post_kernel(const float* __restrict__ qf, const float* __restrict__ kf,
                                const float* __restrict__ vf, const float* __restrict__ ub,
                                const float* __restrict__ vb, half_t* __restrict__ quh,
                                half_t* __restrict__ qvh, half_t* __restrict__ khh,
                                half_t* __restrict__ vth) {
  int idx = blockIdx.x * 256 + threadIdx.x;
  if (idx >= BB * SS * DD) return;
  int c = idx & 127;
  int m = idx >> 7;
  int s = m & (SS - 1);
  int b = m >> 10;
  int h = c >> 5, d = c & 31;
  float q = qf[idx];
  quh[idx] = (half_t)(q + ub[c]);
  qvh[idx] = (half_t)(q + vb[c]);
  khh[idx] = (half_t)kf[idx];
  vth[(((size_t)(b * HH + h)) * DHH + d) * SS + s] = (half_t)vf[idx];
}

// Transformer-XL relative shift gather from staged PS0 (rows i0..i0+16)
static __device__ __forceinline__ float ps_shift(const half_t* ps0, int r, int i, int c) {
  if (c <= i)      return (float)ps0[r * SS + (SS - 1) + c - i];
  if (c == i + 1)  return 0.f;
  return (float)ps0[(r + 1) * SS + (c - i - 2)];
}

// Flash attention with rel-pos shift. 2 waves/block, one 16-row q-block per wave.
__global__ void __launch_bounds__(64)
flash_attn_kernel(const half_t* __restrict__ quh, const half_t* __restrict__ qvh,
                  const half_t* __restrict__ kh, const half_t* __restrict__ ph,
                  const half_t* __restrict__ vth, half_t* __restrict__ ctxh) {
  extern __shared__ char smem_raw[];
  half_t* smem = (half_t*)smem_raw;
  int wave = threadIdx.x >> 5, lane = threadIdx.x & 31;
  half_t* ps0   = smem + (size_t)wave * (17 * SS + 16 * 32);
  half_t* ptile = ps0 + 17 * SS;
  int b = blockIdx.y >> 2, h = blockIdx.y & 3;
  int i0 = (blockIdx.x * 2 + wave) * 16;
  int hi = lane >> 4, cl = lane & 15;

  const half_t* qbase  = quh + ((size_t)(b * SS + i0)) * DD + h * DHH;
  const half_t* qvbase = qvh + ((size_t)(b * SS + i0)) * DD + h * DHH;
  v16h a_qu  = frag_load_a(qbase, DD);
  v16h a_qv  = frag_load_a(qvbase, DD);
  v16h a_qv1 = frag_load_a_guard(qvbase + DD, DD, i0 + 1, SS);  // rows i0+1..i0+16

  v8f zero8;
#pragma unroll
  for (int j = 0; j < 8; ++j) zero8[j] = 0.f;

  // PS0 pre-pass: pos logits for rows i0..i0+16 over all t
  const half_t* pbase = ph + h * DHH;
  for (int t0 = 0; t0 < SS; t0 += 16) {
    v16h bp = frag_load_b(pbase + (size_t)t0 * DD, DD);
    v8f d0 = wmma_f16(a_qv, bp, zero8);
#pragma unroll
    for (int j = 0; j < 8; ++j) ps0[(j + 8 * hi) * SS + t0 + cl] = (half_t)d0[j];
    v8f d1 = wmma_f16(a_qv1, bp, zero8);
    if (hi) ps0[16 * SS + t0 + cl] = (half_t)d1[7];  // row i0+16
  }

  float mrow[8], lrow[8];
  v8f o0, o1;
#pragma unroll
  for (int j = 0; j < 8; ++j) { mrow[j] = -1e30f; lrow[j] = 0.f; o0[j] = 0.f; o1[j] = 0.f; }
  const float scl = 0.088388347648318447f;  // 1/sqrt(128)
  const half_t* kbase = kh + ((size_t)b * SS) * DD + h * DHH;
  const half_t* vbase = vth + ((size_t)(b * HH + h)) * DHH * SS;

  for (int c0 = 0; c0 < SS; c0 += 32) {
    v16h bk0 = frag_load_b(kbase + (size_t)c0 * DD, DD);
    v16h bk1 = frag_load_b(kbase + (size_t)(c0 + 16) * DD, DD);
    v8f sA = wmma_f16(a_qu, bk0, zero8);
    v8f sB = wmma_f16(a_qu, bk1, zero8);
#pragma unroll
    for (int j = 0; j < 8; ++j) {
      int r = j + 8 * hi;
      int row = i0 + r;
      float v0 = (sA[j] + ps_shift(ps0, r, row, c0 + cl)) * scl;
      float v1 = (sB[j] + ps_shift(ps0, r, row, c0 + 16 + cl)) * scl;
      float vm = fmaxf(v0, v1);
#pragma unroll
      for (int off = 1; off < 16; off <<= 1) vm = fmaxf(vm, __shfl_xor(vm, off));
      float mn = fmaxf(mrow[j], vm);
      float rescale = __expf(mrow[j] - mn);
      mrow[j] = mn;
      float p0 = __expf(v0 - mn), p1 = __expf(v1 - mn);
      float rs = p0 + p1;
#pragma unroll
      for (int off = 1; off < 16; off <<= 1) rs += __shfl_xor(rs, off);
      lrow[j] = lrow[j] * rescale + rs;
      o0[j] *= rescale;
      o1[j] *= rescale;
      ptile[r * 32 + cl] = (half_t)p0;
      ptile[r * 32 + 16 + cl] = (half_t)p1;
    }
    v16h ap  = frag_load_a(ptile, 32);
    v16h bv0 = frag_load_b(vbase + c0, SS);
    v16h bv1 = frag_load_b(vbase + (size_t)16 * SS + c0, SS);
    o0 = wmma_f16(ap, bv0, o0);
    o1 = wmma_f16(ap, bv1, o1);
  }
#pragma unroll
  for (int j = 0; j < 8; ++j) {
    float inv = 1.f / lrow[j];
    int row = i0 + j + 8 * hi;
    size_t o = ((size_t)(b * SS + row)) * DD + h * DHH + cl;
    ctxh[o] = (half_t)(o0[j] * inv);
    ctxh[o + 16] = (half_t)(o1[j] * inv);
  }
}

// ---------------------------------------------------------------- conv-module aux
__global__ void __launch_bounds__(256)
jointstats_kernel(const float* __restrict__ y, float* __restrict__ stats) {
  int b = blockIdx.x;
  const float* p = y + (size_t)b * SS * DD;
  float s = 0.f, s2 = 0.f;
  for (int i = threadIdx.x; i < SS * DD; i += 256) {
    float v = p[i]; s += v; s2 += v * v;
  }
  __shared__ float sh1[256], sh2[256];
  sh1[threadIdx.x] = s; sh2[threadIdx.x] = s2;
  __syncthreads();
  for (int st = 128; st > 0; st >>= 1) {
    if (threadIdx.x < st) { sh1[threadIdx.x] += sh1[threadIdx.x + st]; sh2[threadIdx.x] += sh2[threadIdx.x + st]; }
    __syncthreads();
  }
  if (threadIdx.x == 0) {
    float mean = sh1[0] / (float)(SS * DD);
    float var  = sh2[0] / (float)(SS * DD) - mean * mean;
    stats[2 * b] = mean;
    stats[2 * b + 1] = rsqrtf(var + EPSF);
  }
}

__global__ void glu_kernel(const float* __restrict__ z2, float* __restrict__ z3) {
  int idx = blockIdx.x * 256 + threadIdx.x;
  if (idx >= BB * SS * DD) return;
  int c = idx & 127;
  int m = idx >> 7;
  float a = z2[(size_t)m * 256 + c];
  float g = z2[(size_t)m * 256 + 128 + c];
  z3[idx] = a / (1.f + __expf(-g));
}

__global__ void __launch_bounds__(256)
bnstats_kernel(const float* __restrict__ z, float* __restrict__ stats) {
  int c = blockIdx.x;
  float s = 0.f, s2 = 0.f;
  for (int m = threadIdx.x; m < BB * SS; m += 256) {
    float v = z[(size_t)m * DD + c]; s += v; s2 += v * v;
  }
  __shared__ float sh1[256], sh2[256];
  sh1[threadIdx.x] = s; sh2[threadIdx.x] = s2;
  __syncthreads();
  for (int st = 128; st > 0; st >>= 1) {
    if (threadIdx.x < st) { sh1[threadIdx.x] += sh1[threadIdx.x + st]; sh2[threadIdx.x] += sh2[threadIdx.x + st]; }
    __syncthreads();
  }
  if (threadIdx.x == 0) {
    float mean = sh1[0] / (float)(BB * SS);
    float var  = sh2[0] / (float)(BB * SS) - mean * mean;
    stats[2 * c] = mean;
    stats[2 * c + 1] = rsqrtf(var + EPSF);
  }
}

__global__ void bnapply_kernel(const float* __restrict__ z3, const float* __restrict__ stats,
                               const float* __restrict__ gw, const float* __restrict__ gb,
                               half_t* __restrict__ outh) {
  int idx = blockIdx.x * 256 + threadIdx.x;
  if (idx >= BB * SS * DD) return;
  int c = idx & 127;
  outh[idx] = (half_t)((z3[idx] - stats[2 * c]) * stats[2 * c + 1] * gw[c] + gb[c]);
}

// ---------------------------------------------------------------- host
extern "C" void kernel_launch(void* const* d_in, const int* in_sizes, int n_in,
                              void* d_out, int out_size, void* d_ws, size_t ws_size,
                              hipStream_t stream) {
  const float* inputs   = (const float*)d_in[0];
  const float* dcnn_w   = (const float*)d_in[2];
  const float* dcnn_b   = (const float*)d_in[3];
  const float* cm_ln_w  = (const float*)d_in[4];
  const float* cm_ln_b  = (const float*)d_in[5];
  const float* cm_pw1_w = (const float*)d_in[6];
  const float* cm_pw1_b = (const float*)d_in[7];
  const float* cm_dw_w  = (const float*)d_in[8];
  const float* cm_dw_b  = (const float*)d_in[9];
  const float* cm_bn_w  = (const float*)d_in[10];
  const float* cm_bn_b  = (const float*)d_in[11];
  const float* cm_pw2_w = (const float*)d_in[12];
  const float* cm_pw2_b = (const float*)d_in[13];
  const float* ln_attn_w = (const float*)d_in[14];
  const float* ln_attn_b = (const float*)d_in[15];
  const float* wq = (const float*)d_in[16];
  const float* bq = (const float*)d_in[17];
  const float* wk = (const float*)d_in[18];
  const float* bk = (const float*)d_in[19];
  const float* wv = (const float*)d_in[20];
  const float* bv = (const float*)d_in[21];
  const float* wp = (const float*)d_in[22];
  const float* u_bias = (const float*)d_in[23];
  const float* v_bias = (const float*)d_in[24];
  const float* wo = (const float*)d_in[25];
  const float* bo = (const float*)d_in[26];
  const float* ff_ln_w = (const float*)d_in[27];
  const float* ff_ln_b = (const float*)d_in[28];
  const float* ff_w1 = (const float*)d_in[29];
  const float* ff_b1 = (const float*)d_in[30];
  const float* ff_w2 = (const float*)d_in[31];
  const float* ff_b2 = (const float*)d_in[32];
  float* out = (float*)d_out;
  char* ws = (char*)d_ws;

  const size_t MB = 1ull << 20;
  // f16 weight region
  half_t* wq_h  = (half_t*)(ws + 0);
  half_t* wk_h  = (half_t*)(ws + 32768);
  half_t* wv_h  = (half_t*)(ws + 65536);
  half_t* wo_h  = (half_t*)(ws + 98304);
  half_t* wp_h  = (half_t*)(ws + 131072);
  half_t* pw1_h = (half_t*)(ws + 163840);
  half_t* pw2_h = (half_t*)(ws + 229376);
  half_t* ffw1_h = (half_t*)(ws + 262144);
  half_t* ffw2_h = (half_t*)(ws + 393216);
  half_t* dcnn_h = (half_t*)(ws + 524288);
  half_t* dw_h   = (half_t*)(ws + 1540096);
  // persistent activations
  half_t* XLNH = (half_t*)(ws + 8 * MB);
  half_t* CTXH = (half_t*)(ws + 12 * MB);
  float*  X1   = (float*)(ws + 16 * MB);
  half_t* X1H  = (half_t*)(ws + 24 * MB);
  float*  X2   = (float*)(ws + 28 * MB);
  half_t* PEH  = (half_t*)(ws + 36 * MB);
  half_t* PH   = (half_t*)(ws + 36 * MB + 262144);
  float*  JSTATS = (float*)(ws + 36 * MB + 524288);
  float*  BSTATS = (float*)(ws + 36 * MB + 528384);
  // reusable region T = 40MB (attention temps, then conv temps, then FFN temps)
  char* T = ws + 40 * MB;
  float*  QF  = (float*)(T);
  float*  KF  = (float*)(T + 8 * MB);
  float*  VF  = (float*)(T + 16 * MB);
  half_t* QUH = (half_t*)(T + 24 * MB);
  half_t* QVH = (half_t*)(T + 28 * MB);
  half_t* KH  = (half_t*)(T + 32 * MB);
  half_t* VTH = (half_t*)(T + 36 * MB);
  float*  Y   = (float*)(T);             // reuse after attention
  half_t* ZH  = (half_t*)(T + 8 * MB);
  half_t* Z1H = (half_t*)(T + 12 * MB);
  float*  Z2  = (float*)(T + 20 * MB);
  float*  Z3  = (float*)(T + 36 * MB);
  half_t* Z4H = (half_t*)(T + 44 * MB);
  half_t* FH  = (half_t*)(T);            // reuse after conv module
  half_t* F1H = (half_t*)(T + 4 * MB);

  const int M = BB * SS;         // 16384
  const int NEL = M * DD;        // 2097152
  dim3 blk(256);
  dim3 gel((NEL + 255) / 256);

  // ---- pack weights to f16
  f32_to_f16_kernel<<<dim3(64), blk, 0, stream>>>(wq, wq_h, 16384);
  f32_to_f16_kernel<<<dim3(64), blk, 0, stream>>>(wk, wk_h, 16384);
  f32_to_f16_kernel<<<dim3(64), blk, 0, stream>>>(wv, wv_h, 16384);
  f32_to_f16_kernel<<<dim3(64), blk, 0, stream>>>(wo, wo_h, 16384);
  f32_to_f16_kernel<<<dim3(64), blk, 0, stream>>>(wp, wp_h, 16384);
  f32_to_f16_kernel<<<dim3(128), blk, 0, stream>>>(cm_pw1_w, pw1_h, 32768);
  f32_to_f16_kernel<<<dim3(64), blk, 0, stream>>>(cm_pw2_w, pw2_h, 16384);
  f32_to_f16_kernel<<<dim3(256), blk, 0, stream>>>(ff_w1, ffw1_h, 65536);
  f32_to_f16_kernel<<<dim3(256), blk, 0, stream>>>(ff_w2, ffw2_h, 65536);
  pack_conv_kernel<<<dim3((128 * 128 * KT + 255) / 256), blk, 0, stream>>>(dcnn_w, dcnn_h, 128, 128, KT);
  pack_conv_kernel<<<dim3((256 * 256 * KT + 255) / 256), blk, 0, stream>>>(cm_dw_w, dw_h, 256, 256, KT);

  // ---- MHSA
  pe_kernel<<<dim3((SS * DD + 255) / 256), blk, 0, stream>>>(PEH);
  rowln_kernel<<<dim3(M / 8), blk, 0, stream>>>(inputs, ln_attn_w, ln_attn_b, nullptr, XLNH);

  dim3 g128_2(M / 128, 2);
  size_t wsm128 = 64 * 128 * sizeof(half_t);   // 16 KB
  size_t wsm256 = 64 * 256 * sizeof(half_t);   // 32 KB
  size_t wsm512 = 64 * 512 * sizeof(half_t);   // 64 KB
  gemm_conv_kernel<<<g128_2, blk, wsm128, stream>>>(XLNH, wq_h, bq, nullptr, QF, nullptr, SS, 128, 128, 1, 0, 0);
  gemm_conv_kernel<<<g128_2, blk, wsm128, stream>>>(XLNH, wk_h, bk, nullptr, KF, nullptr, SS, 128, 128, 1, 0, 0);
  gemm_conv_kernel<<<g128_2, blk, wsm128, stream>>>(XLNH, wv_h, bv, nullptr, VF, nullptr, SS, 128, 128, 1, 0, 0);
  gemm_conv_kernel<<<dim3(SS / 128, 2), blk, wsm128, stream>>>(PEH, wp_h, nullptr, nullptr, nullptr, PH, SS, 128, 128, 1, 0, 0);

  qkv_post_kernel<<<gel, blk, 0, stream>>>(QF, KF, VF, u_bias, v_bias, QUH, QVH, KH, VTH);

  size_t smem = 2 * (17 * SS + 16 * 32) * sizeof(half_t);  // 71680 B
  flash_attn_kernel<<<dim3(SS / 32, BB * HH), dim3(64), smem, stream>>>(QUH, QVH, KH, PH, VTH, CTXH);

  // x1 = inputs + ctx @ wo^T + bo   (f32 and f16 copies)
  gemm_conv_kernel<<<g128_2, blk, wsm128, stream>>>(CTXH, wo_h, bo, inputs, X1, X1H, SS, 128, 128, 1, 0, 0);

  // ---- ConvModule
  gemm_conv_kernel<<<g128_2, blk, wsm128, stream>>>(X1H, dcnn_h, dcnn_b, nullptr, Y, nullptr, SS, 128, 128, KT, PADK, 0);
  jointstats_kernel<<<dim3(BB), blk, 0, stream>>>(Y, JSTATS);
  rowln_kernel<<<dim3(M / 8), blk, 0, stream>>>(Y, cm_ln_w, cm_ln_b, JSTATS, ZH);
  gemm_conv_kernel<<<dim3(M / 128, 4), blk, wsm128, stream>>>(ZH, pw1_h, cm_pw1_b, nullptr, nullptr, Z1H, SS, 128, 256, 1, 0, 1);
  gemm_conv_kernel<<<dim3(M / 128, 4), blk, wsm256, stream>>>(Z1H, dw_h, cm_dw_b, nullptr, Z2, nullptr, SS, 256, 256, KT, PADK, 0);
  glu_kernel<<<gel, blk, 0, stream>>>(Z2, Z3);
  bnstats_kernel<<<dim3(128), blk, 0, stream>>>(Z3, BSTATS);
  bnapply_kernel<<<gel, blk, 0, stream>>>(Z3, BSTATS, cm_bn_w, cm_bn_b, Z4H);
  gemm_conv_kernel<<<g128_2, blk, wsm128, stream>>>(Z4H, pw2_h, cm_pw2_b, X1, X2, nullptr, SS, 128, 128, 1, 0, 0);

  // ---- FeedForward
  rowln_kernel<<<dim3(M / 8), blk, 0, stream>>>(X2, ff_ln_w, ff_ln_b, nullptr, FH);
  gemm_conv_kernel<<<dim3(M / 128, 8), blk, wsm128, stream>>>(FH, ffw1_h, ff_b1, nullptr, nullptr, F1H, SS, 128, 512, 1, 0, 1);
  gemm_conv_kernel<<<g128_2, blk, wsm512, stream>>>(F1H, ffw2_h, ff_b2, X2, out, nullptr, SS, 512, 128, 1, 0, 0);

  (void)in_sizes; (void)n_in; (void)out_size; (void)ws_size;
}
